// DeformConvBlock_53197464929196
// MI455X (gfx1250) — matrix operand order repaired and checked by reading, not compiled, not run on hardware
//
#include <hip/hip_runtime.h>

typedef __attribute__((ext_vector_type(16))) _Float16 v16h;
typedef __attribute__((ext_vector_type(8)))  _Float16 v8h;
typedef __attribute__((ext_vector_type(8)))  float    v8f;

namespace {
constexpr int kB   = 4;
constexpr int kCI  = 128;
constexpr int kCO  = 128;
constexpr int kH   = 128;
constexpr int kW   = 128;
constexpr int kKK  = 9;                 // 3x3 taps
constexpr int kHW  = kH * kW;           // 16384
constexpr int kNPix = kB * kHW;         // 65536
constexpr int kKtot = kCI * kKK;        // 1152 (GEMM K dimension)
constexpr int kTiles = kKtot / 32;      // 36 K-tiles of 32
constexpr int kMT  = 32;                // pixels per block (two 16-row M tiles)
constexpr float kEps = 1e-3f;
}

// ---------------------------------------------------------------------------
// Kernel 1: zero the global stats accumulators (sum[128], sumsq[128]).
// ---------------------------------------------------------------------------
__global__ void zero_stats_kernel(float* __restrict__ stats) {
  int t = blockIdx.x * blockDim.x + threadIdx.x;
  if (t < 256) stats[t] = 0.f;
}

// ---------------------------------------------------------------------------
// Kernel 2: offset conv, x(4,128,128,128) * w_off(18,128,3,3) + b_off
//           -> off(4,18,128,128). Plain fp32; ~2.7 GFLOP (minor cost).
//           grid = (NPix/64, 18), block = 64.
// ---------------------------------------------------------------------------
__global__ void offset_conv_kernel(const float* __restrict__ x,
                                   const float* __restrict__ w_off,
                                   const float* __restrict__ b_off,
                                   float* __restrict__ off_out) {
  const int P  = blockIdx.x * blockDim.x + threadIdx.x;  // pixel id 0..65535
  const int oc = blockIdx.y;                              // 0..17
  const int b  = P >> 14;
  const int hw = P & (kHW - 1);
  const int h  = hw >> 7;
  const int w  = hw & (kW - 1);
  const float* xb = x + (size_t)b * kCI * kHW;
  float acc = b_off[oc];
  for (int ci = 0; ci < kCI; ++ci) {
    const float* xp = xb + ci * kHW;
    const float* wp = w_off + (oc * kCI + ci) * kKK;   // uniform -> s_loads
#pragma unroll
    for (int ky = 0; ky < 3; ++ky) {
      const int y = h - 1 + ky;
      if ((unsigned)y < (unsigned)kH) {
        const float* row = xp + y * kW;
#pragma unroll
        for (int kx = 0; kx < 3; ++kx) {
          const int xx = w - 1 + kx;
          if ((unsigned)xx < (unsigned)kW)
            acc = fmaf(row[xx], wp[ky * 3 + kx], acc);
        }
      }
    }
  }
  off_out[(b * 18 + oc) * kHW + hw] = acc;
}

// ---------------------------------------------------------------------------
// Kernel 3: pack w_dcn (CO,CI,3,3) fp32 -> f16 in the EXACT per-lane WMMA
// B-fragment layout: wfrag[((tile*8 + wave)*32 + lane)*16 + i].
// Each lane then fetches its whole fragment as 32 contiguous bytes.
// tile = kk*4 + cb/32 ; lane -> (kb0, n) per ISA 16-bit B 32x16 layout.
// ---------------------------------------------------------------------------
__global__ void pack_w_kernel(const float* __restrict__ w_dcn,
                              _Float16* __restrict__ wfrag) {
  int e = blockIdx.x * blockDim.x + threadIdx.x;   // over 36*8*32*16 = 147456
  if (e >= kTiles * 8 * 32 * 16) return;
  const int i    = e & 15;
  const int lane = (e >> 4) & 31;
  const int wave = (e >> 9) & 7;
  const int tile = e >> 12;                 // 0..35
  const int kk   = tile >> 2;               // tap 0..8
  const int cb   = (tile & 3) * 32;         // channel tile base
  const int kb0  = (lane < 16) ? 0 : 8;     // B k-offset within 32-tile
  const int n    = lane & 15;
  const int c    = cb + kb0 + (i & 7) + ((i >> 3) << 4);  // input channel
  const int co   = wave * 16 + n;           // output channel
  wfrag[e] = (_Float16)w_dcn[(co * kCI + c) * kKK + kk];
}

// ---------------------------------------------------------------------------
// Kernel 4: fused bilinear deformable sampling + WMMA GEMM + BN stats.
// One block (256 thr = 8 wave32) computes 32 pixels x all 128 out-channels.
// grid = NPix/32 = 2048 blocks.
// GEMM: M=32 (two 16-row tiles), N=128 (8 waves x 16), K=1152 in 36 tiles
// of 32 using v_wmma_f32_16x16x32_f16 with fp32 accumulation.
// Per K-tile: one barrier feeds TWO WMMAs; the B fragment (two coalesced
// global_load_b128) is reused in registers for both. A-stage LDS is
// pixel-major+padded so each A fragment is two ds_load_b128; s_A is
// double-buffered so there is a single barrier per K-tile.
// ---------------------------------------------------------------------------
__global__ __launch_bounds__(256) void
deform_wmma_kernel(const float* __restrict__ x,
                   const float* __restrict__ off_ws,
                   const _Float16* __restrict__ wfrag,
                   float* __restrict__ raw,
                   float* __restrict__ g_sum,
                   float* __restrict__ g_sq) {
  __shared__ __align__(16) _Float16 s_A[2][kMT][40];   // [buf][pixel][K], 80B rows
  __shared__ __align__(16) float    s_out[8][16][kMT]; // [wave][n][pixel]
  __shared__ float                  s_sum[128];
  __shared__ float                  s_sq[128];

  const int tid  = threadIdx.x;
  const int wave = tid >> 5;
  const int lane = tid & 31;

  if (tid < 128) { s_sum[tid] = 0.f; s_sq[tid] = 0.f; }

  // sampling-phase identity: pixel p (0..31), channel-row cr (0..7)
  const int p  = tid & 31;
  const int cr = tid >> 5;                    // 0..7 -> channels cr+{0,8,16,24}
  const int Pp = blockIdx.x * kMT + p;
  const int bp = Pp >> 14;
  const int hwp = Pp & (kHW - 1);
  const int hp = hwp >> 7;
  const int wpix = hwp & (kW - 1);
  const float* xb0 = x + (size_t)(bp * kCI + cr) * kHW;

  // wmma-phase identity (ISA 7.12.2 layouts, wave32)
  const int m  = lane & 15;                   // A row within 16-row tile
  const int hk = (lane < 16) ? 0 : 8;         // A k-offset within tile
  const int ncol = wave * 16 + (lane & 15);   // output channel column

  v8f acc0 = {};   // pixels 0..15
  v8f acc1 = {};   // pixels 16..31

  for (int kk = 0; kk < kKK; ++kk) {
    // --- per-thread bilinear geometry for pixel p, tap kk (registers) -----
    const int ky = kk / 3, kx = kk % 3;
    const float dy = off_ws[(bp * 18 + kk * 2) * kHW + hwp];
    const float dx = off_ws[(bp * 18 + kk * 2 + 1) * kHW + hwp];
    const float ysf = (float)(hp - 1 + ky) + dy;
    const float xsf = (float)(wpix - 1 + kx) + dx;
    const float y0f = floorf(ysf), x0f = floorf(xsf);
    const float fy = ysf - y0f, fx = xsf - x0f;
    const int y0 = (int)y0f, x0 = (int)x0f;
    int   ci4[4];
    float cw4[4];
    {
      const float cw[4] = { (1.f - fy) * (1.f - fx), (1.f - fy) * fx,
                            fy * (1.f - fx),         fy * fx };
#pragma unroll
      for (int c4 = 0; c4 < 4; ++c4) {
        const int yi = y0 + (c4 >> 1);
        const int xi = x0 + (c4 & 1);
        const bool valid = (yi >= 0) && (yi < kH) && (xi >= 0) && (xi < kW);
        const int yc = min(max(yi, 0), kH - 1);
        const int xc = min(max(xi, 0), kW - 1);
        ci4[c4] = yc * kW + xc;
        cw4[c4] = valid ? cw[c4] : 0.f;
      }
    }

#pragma unroll
    for (int cbi = 0; cbi < 4; ++cbi) {
      const int tile = kk * 4 + cbi;
      const int buf  = tile & 1;

      // --- gather + bilinear: 32 px x 32 ch into LDS A-stage (f16) --------
      {
        const float* xq = xb0 + cbi * 32 * kHW;
#pragma unroll
        for (int cc = 0; cc < 4; ++cc) {      // channels cr + cc*8
          const float* xp = xq + cc * 8 * kHW;
          const float v = cw4[0] * xp[ci4[0]] + cw4[1] * xp[ci4[1]] +
                          cw4[2] * xp[ci4[2]] + cw4[3] * xp[ci4[3]];
          s_A[buf][p][cr + cc * 8] = (_Float16)v;
        }
      }
      __syncthreads();   // single barrier per K-tile (double-buffered s_A)

      // --- B fragment: 32 contiguous bytes per lane, reused by 2 WMMAs ----
      const _Float16* wpk = wfrag + ((size_t)((tile * 8 + wave) * 32 + lane) << 4);
      const v8h blo = *(const v8h*)wpk;
      const v8h bhi = *(const v8h*)(wpk + 8);
      v16h bfrag;
#pragma unroll
      for (int i = 0; i < 8; ++i) { bfrag[i] = blo[i]; bfrag[8 + i] = bhi[i]; }

      // --- A fragment tile 0 (pixels 0..15): two 16B LDS reads ------------
      {
        const v8h alo = *(const v8h*)(&s_A[buf][m][hk]);
        const v8h ahi = *(const v8h*)(&s_A[buf][m][hk + 16]);
        v16h afrag;
#pragma unroll
        for (int i = 0; i < 8; ++i) { afrag[i] = alo[i]; afrag[8 + i] = ahi[i]; }
        acc0 = __builtin_amdgcn_wmma_f32_16x16x32_f16(
            false, afrag, false, bfrag, (short)0, acc0, false, false);
      }
      // --- A fragment tile 1 (pixels 16..31) ------------------------------
      {
        const v8h alo = *(const v8h*)(&s_A[buf][m + 16][hk]);
        const v8h ahi = *(const v8h*)(&s_A[buf][m + 16][hk + 16]);
        v16h afrag;
#pragma unroll
        for (int i = 0; i < 8; ++i) { afrag[i] = alo[i]; afrag[8 + i] = ahi[i]; }
        acc1 = __builtin_amdgcn_wmma_f32_16x16x32_f16(
            false, afrag, false, bfrag, (short)0, acc1, false, false);
      }
    }
  }

  // --- transpose wave tiles through LDS + per-channel BN stats ------------
  const int m0 = (lane < 16) ? 0 : 8;
  const int n16 = lane & 15;
  float s = 0.f, q = 0.f;
#pragma unroll
  for (int r = 0; r < 8; ++r) {
    const float v0 = acc0[r];
    const float v1 = acc1[r];
    s_out[wave][n16][r + m0]      = v0;
    s_out[wave][n16][16 + r + m0] = v1;
    s += v0 + v1;
    q += v0 * v0 + v1 * v1;
  }
  atomicAdd(&s_sum[ncol], s);   // ds_add_f32
  atomicAdd(&s_sq[ncol],  q);
  __syncthreads();

  // coalesced writeback: 2 lanes per output row, four float4 each (128B/row)
  {
    const int row  = lane >> 1;         // 0..15  (n within wave tile)
    const int part = lane & 1;          // 0..1 -> pixels part*16..part*16+15
    const int co   = wave * 16 + row;
    const int Pb   = blockIdx.x * kMT;  // 32 pixels share b,h; w contiguous
    const int b2   = Pb >> 14;
    const int hw2  = Pb & (kHW - 1);
    float* dst = raw + ((size_t)(b2 * kCO + co)) * kHW + hw2 + part * 16;
    const float* srcp = &s_out[wave][row][part * 16];
#pragma unroll
    for (int v = 0; v < 4; ++v)
      *(float4*)(dst + v * 4) = *(const float4*)(srcp + v * 4);
  }

  if (tid < 128) {
    atomicAdd(&g_sum[tid], s_sum[tid]);   // global_atomic_add_f32
    atomicAdd(&g_sq[tid],  s_sq[tid]);
  }
}

// ---------------------------------------------------------------------------
// Kernel 5: finalize BN stats -> per-channel scale/shift.
// ---------------------------------------------------------------------------
__global__ void finalize_stats_kernel(const float* __restrict__ g_sum,
                                      const float* __restrict__ g_sq,
                                      const float* __restrict__ gamma,
                                      const float* __restrict__ beta,
                                      float* __restrict__ scale,
                                      float* __restrict__ shift) {
  int c = threadIdx.x;
  if (c < kCO) {
    const float invN = 1.f / (float)kNPix;
    const float mean = g_sum[c] * invN;
    const float var  = g_sq[c] * invN - mean * mean;
    const float sc   = gamma[c] * rsqrtf(var + kEps);
    scale[c] = sc;
    shift[c] = beta[c] - mean * sc;
  }
}

// ---------------------------------------------------------------------------
// Kernel 6: y = relu(raw*scale + shift), float4 vectorized.
// ---------------------------------------------------------------------------
__global__ void bn_relu_kernel(const float* __restrict__ raw,
                               const float* __restrict__ scale,
                               const float* __restrict__ shift,
                               float* __restrict__ out) {
  const int gid = blockIdx.x * blockDim.x + threadIdx.x;   // float4 index
  const int idx = gid << 2;
  const int co  = (idx >> 14) & (kCO - 1);
  float4 v = ((const float4*)raw)[gid];
  const float sc = scale[co], sh = shift[co];
  v.x = fmaxf(fmaf(v.x, sc, sh), 0.f);
  v.y = fmaxf(fmaf(v.y, sc, sh), 0.f);
  v.z = fmaxf(fmaf(v.z, sc, sh), 0.f);
  v.w = fmaxf(fmaf(v.w, sc, sh), 0.f);
  ((float4*)out)[gid] = v;
}

// ---------------------------------------------------------------------------
extern "C" void kernel_launch(void* const* d_in, const int* in_sizes, int n_in,
                              void* d_out, int out_size, void* d_ws, size_t ws_size,
                              hipStream_t stream) {
  const float* x     = (const float*)d_in[0];   // (4,128,128,128)
  const float* w_off = (const float*)d_in[1];   // (18,128,3,3)
  const float* b_off = (const float*)d_in[2];   // (18,)
  const float* w_dcn = (const float*)d_in[3];   // (128,128,3,3)
  const float* gamma = (const float*)d_in[4];   // (128,)
  const float* beta  = (const float*)d_in[5];   // (128,)
  float* out = (float*)d_out;                   // (4,128,128,128)

  // workspace layout (floats): offsets | raw conv out | stats | packed f16 W
  float* off_ws = (float*)d_ws;                       // 4*18*16384 = 1,179,648
  float* raw    = off_ws + (size_t)kB * 18 * kHW;     // 8,388,608
  float* stats  = raw + (size_t)kB * kCO * kHW;       // 512 floats
  float* g_sum  = stats;
  float* g_sq   = stats + 128;
  float* scale  = stats + 256;
  float* shift  = stats + 384;
  _Float16* wfrag = (_Float16*)(stats + 512);         // 147,456 halves

  zero_stats_kernel<<<1, 256, 0, stream>>>(stats);

  offset_conv_kernel<<<dim3(kNPix / 64, 18), 64, 0, stream>>>(x, w_off, b_off, off_ws);

  pack_w_kernel<<<(kTiles * 8 * 32 * 16 + 255) / 256, 256, 0, stream>>>(w_dcn, wfrag);

  deform_wmma_kernel<<<kNPix / kMT, 256, 0, stream>>>(x, off_ws, wfrag, raw, g_sum, g_sq);

  finalize_stats_kernel<<<1, 128, 0, stream>>>(g_sum, g_sq, gamma, beta, scale, shift);

  bn_relu_kernel<<<(kB * kCO * kHW / 4) / 256, 256, 0, stream>>>(raw, scale, shift, out);
}
